// FlowNet2C_67619965108277
// MI455X (gfx1250) — matrix operand order repaired and checked by reading, not compiled, not run on hardware
//
#include <hip/hip_runtime.h>
#include <hip/hip_bf16.h>

typedef __attribute__((ext_vector_type(16))) _Float16 v16h;
typedef __attribute__((ext_vector_type(8)))  float    v8f;

// ---------------------------------------------------------------------------
// Generic implicit-GEMM conv / transposed-conv via V_WMMA_F32_16X16X32_F16.
// GEMM view:  Y[Cout, N*Ho*Wo] = W[Cout, Cin*KH*KW] x im2col(X)[Cin*KH*KW, P]
// One wave -> 16(Cout) x 64(pixel) strip = 4 WMMA accumulators sharing one
// A fragment; 4 waves/block share the 16x32 A (weight) tile in LDS (f32).
// Weight tile staging uses gfx1250 GLOBAL_LOAD_ASYNC_TO_LDS (ASYNCcnt) when
// the tile is fully in-bounds; guarded scalar staging otherwise (zero fill).
// KH/KW/MODE are template constants so all k->(ci,kh,kw) math strength-reduces.
// MODE 0: normal conv, weights OIHW (k-contiguous rows).
// MODE 1: transposed conv (lhs_dilation=2, pad=2, k=4), weights stored IOHW,
//         kernel spatially flipped (matches jnp.flip(w).transpose(1,0,2,3)).
// ---------------------------------------------------------------------------
struct ConvParams {
    const float* x;   // input activations (f32), tensor has Cx channels
    const float* w;   // weights (f32)
    const float* b;   // bias (f32), Cout entries
    float*       y;   // output tensor with Cy channels; write at co_off
    int N, Cin, Cx, H, W;
    int Cout, Cy, co_off;
    int stride, pad;
    int Hout, Wout;
    int relu;         // 1 -> leaky relu (0.1)
};

template <int KH, int KW, int MODE>
__device__ __forceinline__ const float* weight_elem_ptr(const ConvParams& p,
                                                        int co, int k) {
    constexpr int KHW = KH * KW;
    const int ci = k / KHW;
    const int kr = k - ci * KHW;
    const int kh = kr / KW;
    const int kw = kr - kh * KW;
    if (MODE == 0) {
        return &p.w[((co * p.Cin + ci) * KH + kh) * KW + kw];
    } else {
        return &p.w[((ci * p.Cout + co) * KH + (KH - 1 - kh)) * KW + (KW - 1 - kw)];
    }
}

template <int KH, int KW, int MODE>
__global__ void __launch_bounds__(128) conv_wmma_kernel(ConvParams p) {
    constexpr int KHW = KH * KW;
    const int lane = threadIdx.x & 31;
    const int wave = threadIdx.x >> 5;
    const int mbase   = blockIdx.y * 16;                   // Cout tile base
    const int pixbase = (blockIdx.x * 4 + wave) * 64;      // 64 pixels per wave
    const int Ktot = p.Cin * KHW;
    const int P    = p.N * p.Hout * p.Wout;

    __shared__ float lds_a[16 * 32];                       // [m][k] f32 tile

    // Decode the 4 pixel groups this wave owns (B/D: N = lane % 16).
    int pn[4], py[4], px[4];
    bool pv[4];
#pragma unroll
    for (int j = 0; j < 4; ++j) {
        const int pix = pixbase + j * 16 + (lane & 15);
        pv[j] = pix < P;
        const int t = pv[j] ? pix : 0;
        px[j] = t % p.Wout;
        py[j] = (t / p.Wout) % p.Hout;
        pn[j] = t / (p.Wout * p.Hout);
    }

    v8f acc[4] = {{}, {}, {}, {}};

    const int am   = lane & 15;                 // A: M = lane % 16
    const int kgrp = (lane >> 4) * 8;           // unused: A also uses kb (lanes 16-31 hold K=16..31)
    const int kb   = (lane >> 4) * 16;          // B: lanes 16-31 hold K=16..31

    const bool mFull       = (mbase + 16) <= p.Cout;
    const bool kContig128  = (MODE == 0) && ((Ktot & 3) == 0);

    for (int k0 = 0; k0 < Ktot; k0 += 32) {
        __syncthreads();
        const bool tileFull = mFull && (k0 + 32) <= Ktot;  // block-uniform

        if (tileFull && kContig128) {
            // ---- async DMA: 16 rows x 128B, one b128 per thread ----
            const int row   = threadIdx.x >> 3;            // 0..15
            const int chunk = (threadIdx.x & 7) * 4;       // element offset
            const float* wp = p.w + (size_t)(mbase + row) * Ktot + k0 + chunk;
            const unsigned ldsoff =
                (unsigned)(size_t)&lds_a[row * 32 + chunk];
            asm volatile("global_load_async_to_lds_b128 %0, %1, off"
                         :: "v"(ldsoff),
                            "v"((unsigned long long)(size_t)wp)
                         : "memory");
            asm volatile("s_wait_asynccnt 0x0" ::: "memory");
        } else if (tileFull) {
            // ---- async DMA: per-element b32 gather ----
            for (int i = threadIdx.x; i < 512; i += 128) {
                const int m  = i >> 5;
                const int kk = i & 31;
                const float* wp =
                    weight_elem_ptr<KH, KW, MODE>(p, mbase + m, k0 + kk);
                const unsigned ldsoff = (unsigned)(size_t)&lds_a[i];
                asm volatile("global_load_async_to_lds_b32 %0, %1, off"
                             :: "v"(ldsoff),
                                "v"((unsigned long long)(size_t)wp)
                             : "memory");
            }
            asm volatile("s_wait_asynccnt 0x0" ::: "memory");
        } else {
            // ---- guarded scalar staging with zero fill (edge tiles) ----
            for (int i = threadIdx.x; i < 512; i += 128) {
                const int m  = i >> 5;
                const int kk = i & 31;
                const int k  = k0 + kk;
                const int co = mbase + m;
                float wv = 0.0f;
                if (co < p.Cout && k < Ktot) {
                    const float* wp = weight_elem_ptr<KH, KW, MODE>(p, co, k);
                    wv = *wp;
                    if (MODE == 0) __builtin_prefetch(wp + 32, 0, 1);
                }
                lds_a[i] = wv;
            }
        }
        __syncthreads();

        // ---- A fragment from LDS (16-bit A 16x32 layout), f32 -> f16 ----
        v16h afrag;
#pragma unroll
        for (int e = 0; e < 16; ++e) {
            const int g = e >> 3;                 // VGPR group 0..3 / 4..7
            const int o = e & 7;
            afrag[e] = (_Float16)lds_a[am * 32 + kb + e];
        }

        // ---- incremental k -> (ci, kh, kw) decomposition, shared by all j ----
        int cie[16], khe[16], kwe[16];
        bool kva[16];
        {
            const int kstart = k0 + kb;
            int ci = kstart / KHW;
            int kr = kstart - ci * KHW;
            int kh = kr / KW;
            int kw = kr - kh * KW;
#pragma unroll
            for (int e = 0; e < 16; ++e) {
                cie[e] = ci; khe[e] = kh; kwe[e] = kw;
                kva[e] = (kstart + e) < Ktot;
                if (++kw == KW) { kw = 0; if (++kh == KH) { kh = 0; ++ci; } }
            }
        }

        // ---- 4 B fragments (im2col gather) -> 4 WMMA issues on shared A ----
#pragma unroll
        for (int j = 0; j < 4; ++j) {
            v16h bfrag;
#pragma unroll
            for (int e = 0; e < 16; ++e) {
                float xv = 0.0f;
                if (pv[j] && kva[e]) {
                    if (MODE == 0) {
                        const int iy = py[j] * p.stride - p.pad + khe[e];
                        const int ix = px[j] * p.stride - p.pad + kwe[e];
                        if (iy >= 0 && iy < p.H && ix >= 0 && ix < p.W)
                            xv = p.x[((pn[j] * p.Cx + cie[e]) * p.H + iy) * p.W + ix];
                    } else {
                        // lhs_dilation = 2, pad = 2 in dilated space
                        const int q = py[j] - 2 + khe[e];
                        const int r = px[j] - 2 + kwe[e];
                        if (q >= 0 && r >= 0 && !(q & 1) && !(r & 1)) {
                            const int iy = q >> 1;
                            const int ix = r >> 1;
                            if (iy < p.H && ix < p.W)
                                xv = p.x[((pn[j] * p.Cx + cie[e]) * p.H + iy) * p.W + ix];
                        }
                    }
                }
                bfrag[e] = (_Float16)xv;
            }
            // All 32 lanes reach here every iteration (EXEC all ones).
            acc[j] = __builtin_amdgcn_wmma_f32_16x16x32_f16(
                false, afrag, false, bfrag, (short)0, acc[j], false, false);
        }
    }

    // ---- epilogue: bias + optional leaky relu, guarded stores ----
    const int mofs = (lane >= 16) ? 8 : 0;
#pragma unroll
    for (int j = 0; j < 4; ++j) {
        if (pv[j]) {
#pragma unroll
            for (int v = 0; v < 8; ++v) {
                const int co = mbase + v + mofs;
                if (co < p.Cout) {
                    float r = acc[j][v] + p.b[co];
                    if (p.relu && r < 0.0f) r *= 0.1f;
                    p.y[((pn[j] * p.Cy + p.co_off + co) * p.Hout + py[j]) * p.Wout
                        + px[j]] = r;
                }
            }
        }
    }
}

// ---------------------------------------------------------------------------
// Per-(n,c) mean over the 2 frames: inputs (4,3,2,384,512)
// ---------------------------------------------------------------------------
__global__ void mean_kernel(const float* in, float* mean) {
    const int nc = blockIdx.x;                  // 0..11
    const int cnt = 2 * 384 * 512;
    const float* base = in + (size_t)nc * cnt;
    __shared__ float sdata[256];
    float s = 0.0f;
    for (int i = threadIdx.x; i < cnt; i += 256) s += base[i];
    sdata[threadIdx.x] = s;
    __syncthreads();
    for (int k = 128; k > 0; k >>= 1) {
        if (threadIdx.x < k) sdata[threadIdx.x] += sdata[threadIdx.x + k];
        __syncthreads();
    }
    if (threadIdx.x == 0) mean[nc] = sdata[0] / (float)cnt;
}

// x1/x2 = (frame - mean) / 255
__global__ void preproc_kernel(const float* in, const float* mean,
                               float* x1, float* x2, int total) {
    const int idx = blockIdx.x * blockDim.x + threadIdx.x;
    if (idx >= total) return;
    const int HW = 384 * 512;
    const int rem = idx % HW;
    const int nc  = idx / HW;                   // n*3 + c
    const float m = mean[nc];
    const size_t base = (size_t)nc * 2 * HW;
    x1[idx] = (in[base + rem]      - m) * (1.0f / 255.0f);
    x2[idx] = (in[base + HW + rem] - m) * (1.0f / 255.0f);
}

// ---------------------------------------------------------------------------
// Correlation: out[n,d,y,x] = lrelu( mean_c f1[n,c,y,x]*f2[n,c,y+dy,x+dx] )
// 21x21 displacements, stride 2, written at channel offset 32 of 473-ch buf.
// ---------------------------------------------------------------------------
__global__ void corr_kernel(const float* f1, const float* f2, float* out) {
    const int H = 48, W = 64, C = 256, D = 441;
    const int idx = blockIdx.x * blockDim.x + threadIdx.x;
    const int total = 4 * D * H * W;
    if (idx >= total) return;
    const int x = idx % W;
    const int y = (idx / W) % H;
    const int d = (idx / (W * H)) % D;
    const int n = idx / (W * H * D);
    const int dy = (d / 21) * 2 - 20;
    const int dx = (d % 21) * 2 - 20;
    const int yy = y + dy, xx = x + dx;
    float s = 0.0f;
    if (yy >= 0 && yy < H && xx >= 0 && xx < W) {
        const float* a = f1 + ((size_t)(n * C) * H + y) * W + x;
        const float* b = f2 + ((size_t)(n * C) * H + yy) * W + xx;
        for (int c = 0; c < C; ++c)
            s += a[(size_t)c * H * W] * b[(size_t)c * H * W];
        s *= (1.0f / (float)C);
    }
    if (s < 0.0f) s *= 0.1f;
    out[(((size_t)n * 473 + 32 + d) * H + y) * W + x] = s;
}

// ---------------------------------------------------------------------------
// Bilinear x4 upsample (half-pixel centers, edge clamp) with scale multiply.
// ---------------------------------------------------------------------------
__global__ void upsample_kernel(const float* src, float* dst,
                                int N, int C, int Hs, int Ws, int Hd, int Wd,
                                float mul) {
    const int idx = blockIdx.x * blockDim.x + threadIdx.x;
    const int total = N * C * Hd * Wd;
    if (idx >= total) return;
    const int x = idx % Wd;
    const int y = (idx / Wd) % Hd;
    const int c = (idx / (Wd * Hd)) % C;
    const int n = idx / (Wd * Hd * C);
    const float sy = (y + 0.5f) * 0.25f - 0.5f;
    const float sx = (x + 0.5f) * 0.25f - 0.5f;
    int y0 = (int)floorf(sy), x0 = (int)floorf(sx);
    const float fy = sy - (float)y0, fx = sx - (float)x0;
    int y1 = y0 + 1, x1 = x0 + 1;
    y0 = min(max(y0, 0), Hs - 1); y1 = min(max(y1, 0), Hs - 1);
    x0 = min(max(x0, 0), Ws - 1); x1 = min(max(x1, 0), Ws - 1);
    const float* s = src + ((size_t)(n * C + c)) * Hs * Ws;
    const float v = (1.0f - fy) * ((1.0f - fx) * s[y0 * Ws + x0] + fx * s[y0 * Ws + x1])
                  +         fy  * ((1.0f - fx) * s[y1 * Ws + x0] + fx * s[y1 * Ws + x1]);
    dst[idx] = v * mul;
}

// ---------------------------------------------------------------------------
// Host-side orchestration
// ---------------------------------------------------------------------------
static void launch_conv(hipStream_t s,
                        const float* x, const float* w, const float* b, float* y,
                        int N, int Cin, int Cx, int H, int W,
                        int Cout, int Cy, int co_off,
                        int KH, int KW, int stride, int pad,
                        int Hout, int Wout, int relu, int mode) {
    ConvParams p;
    p.x = x; p.w = w; p.b = b; p.y = y;
    p.N = N; p.Cin = Cin; p.Cx = Cx; p.H = H; p.W = W;
    p.Cout = Cout; p.Cy = Cy; p.co_off = co_off;
    p.stride = stride; p.pad = pad;
    p.Hout = Hout; p.Wout = Wout; p.relu = relu;
    const int P = N * Hout * Wout;
    dim3 grid((P + 255) / 256, (Cout + 15) / 16);
    dim3 block(128);
    if (mode == 1) {
        conv_wmma_kernel<4, 4, 1><<<grid, block, 0, s>>>(p);
    } else if (KH == 7) {
        conv_wmma_kernel<7, 7, 0><<<grid, block, 0, s>>>(p);
    } else if (KH == 5) {
        conv_wmma_kernel<5, 5, 0><<<grid, block, 0, s>>>(p);
    } else if (KH == 3) {
        conv_wmma_kernel<3, 3, 0><<<grid, block, 0, s>>>(p);
    } else {
        conv_wmma_kernel<1, 1, 0><<<grid, block, 0, s>>>(p);
    }
}

extern "C" void kernel_launch(void* const* d_in, const int* in_sizes, int n_in,
                              void* d_out, int out_size, void* d_ws, size_t ws_size,
                              hipStream_t stream) {
    const float* inputs = (const float*)d_in[0];
    const float* c1_w = (const float*)d_in[1];   const float* c1_b = (const float*)d_in[2];
    const float* c2_w = (const float*)d_in[3];   const float* c2_b = (const float*)d_in[4];
    const float* c3_w = (const float*)d_in[5];   const float* c3_b = (const float*)d_in[6];
    const float* cr_w = (const float*)d_in[7];   const float* cr_b = (const float*)d_in[8];
    const float* c31_w = (const float*)d_in[9];  const float* c31_b = (const float*)d_in[10];
    const float* c4_w = (const float*)d_in[11];  const float* c4_b = (const float*)d_in[12];
    const float* c41_w = (const float*)d_in[13]; const float* c41_b = (const float*)d_in[14];
    const float* c5_w = (const float*)d_in[15];  const float* c5_b = (const float*)d_in[16];
    const float* c51_w = (const float*)d_in[17]; const float* c51_b = (const float*)d_in[18];
    const float* c6_w = (const float*)d_in[19];  const float* c6_b = (const float*)d_in[20];
    const float* c61_w = (const float*)d_in[21]; const float* c61_b = (const float*)d_in[22];
    const float* d5_w = (const float*)d_in[23];  const float* d5_b = (const float*)d_in[24];
    const float* d4_w = (const float*)d_in[25];  const float* d4_b = (const float*)d_in[26];
    const float* d3_w = (const float*)d_in[27];  const float* d3_b = (const float*)d_in[28];
    const float* d2_w = (const float*)d_in[29];  const float* d2_b = (const float*)d_in[30];
    const float* pf6_w = (const float*)d_in[31]; const float* pf6_b = (const float*)d_in[32];
    const float* pf5_w = (const float*)d_in[33]; const float* pf5_b = (const float*)d_in[34];
    const float* pf4_w = (const float*)d_in[35]; const float* pf4_b = (const float*)d_in[36];
    const float* pf3_w = (const float*)d_in[37]; const float* pf3_b = (const float*)d_in[38];
    const float* pf2_w = (const float*)d_in[39]; const float* pf2_b = (const float*)d_in[40];
    const float* u6_w = (const float*)d_in[41];  const float* u6_b = (const float*)d_in[42];
    const float* u5_w = (const float*)d_in[43];  const float* u5_b = (const float*)d_in[44];
    const float* u4_w = (const float*)d_in[45];  const float* u4_b = (const float*)d_in[46];
    const float* u3_w = (const float*)d_in[47];  const float* u3_b = (const float*)d_in[48];

    float* ws = (float*)d_ws;
    size_t off = 0;
    auto alloc = [&](size_t n) -> float* {
        float* p = ws + off;
        off += (n + 15) & ~((size_t)15);
        return p;
    };

    float* mean  = alloc(12);
    float* x1    = alloc((size_t)4 * 3 * 384 * 512);
    float* x2    = alloc((size_t)4 * 3 * 384 * 512);
    float* o1a   = alloc((size_t)4 * 64 * 192 * 256);
    float* o1b   = alloc((size_t)4 * 64 * 192 * 256);
    float* o2b   = alloc((size_t)4 * 128 * 96 * 128);
    float* o3a   = alloc((size_t)4 * 256 * 48 * 64);
    float* o3b   = alloc((size_t)4 * 256 * 48 * 64);
    float* cat31 = alloc((size_t)4 * 473 * 48 * 64);   // [redir(32) | corr(441)]
    float* t4    = alloc((size_t)4 * 512 * 24 * 32);
    float* t5    = alloc((size_t)4 * 512 * 12 * 16);
    float* t6    = alloc((size_t)4 * 1024 * 6 * 8);
    float* o6    = alloc((size_t)4 * 1024 * 6 * 8);
    float* flow6 = alloc((size_t)4 * 2 * 6 * 8);
    float* cat5  = alloc((size_t)4 * 1026 * 12 * 16);  // [o5(512) | dc5(512) | uf6(2)]
    float* flow5 = alloc((size_t)4 * 2 * 12 * 16);
    float* cat4  = alloc((size_t)4 * 770 * 24 * 32);   // [o4(512) | dc4(256) | uf5(2)]
    float* flow4 = alloc((size_t)4 * 2 * 24 * 32);
    float* cat3  = alloc((size_t)4 * 386 * 48 * 64);   // [o31(256) | dc3(128) | uf4(2)]
    float* flow3 = alloc((size_t)4 * 2 * 48 * 64);
    float* cat2  = alloc((size_t)4 * 194 * 96 * 128);  // [o2a(128) | dc2(64) | uf3(2)]
    float* flow2 = alloc((size_t)4 * 2 * 96 * 128);

    // --- preprocess ---
    mean_kernel<<<12, 256, 0, stream>>>(inputs, mean);
    {
        const int total = 4 * 3 * 384 * 512;
        preproc_kernel<<<(total + 255) / 256, 256, 0, stream>>>(inputs, mean, x1, x2, total);
    }

    // --- shared encoder ---
    launch_conv(stream, x1, c1_w, c1_b, o1a, 4, 3, 3, 384, 512, 64, 64, 0, 7, 7, 2, 3, 192, 256, 1, 0);
    launch_conv(stream, o1a, c2_w, c2_b, cat2, 4, 64, 64, 192, 256, 128, 194, 0, 5, 5, 2, 2, 96, 128, 1, 0);   // o2a
    launch_conv(stream, cat2, c3_w, c3_b, o3a, 4, 128, 194, 96, 128, 256, 256, 0, 5, 5, 2, 2, 48, 64, 1, 0);
    launch_conv(stream, x2, c1_w, c1_b, o1b, 4, 3, 3, 384, 512, 64, 64, 0, 7, 7, 2, 3, 192, 256, 1, 0);
    launch_conv(stream, o1b, c2_w, c2_b, o2b, 4, 64, 64, 192, 256, 128, 128, 0, 5, 5, 2, 2, 96, 128, 1, 0);
    launch_conv(stream, o2b, c3_w, c3_b, o3b, 4, 128, 128, 96, 128, 256, 256, 0, 5, 5, 2, 2, 48, 64, 1, 0);

    // --- correlation + redirect ---
    {
        const int total = 4 * 441 * 48 * 64;
        corr_kernel<<<(total + 255) / 256, 256, 0, stream>>>(o3a, o3b, cat31);
    }
    launch_conv(stream, o3a, cr_w, cr_b, cat31, 4, 256, 256, 48, 64, 32, 473, 0, 1, 1, 1, 0, 48, 64, 1, 0);

    // --- contracting path ---
    launch_conv(stream, cat31, c31_w, c31_b, cat3, 4, 473, 473, 48, 64, 256, 386, 0, 3, 3, 1, 1, 48, 64, 1, 0); // o31
    launch_conv(stream, cat3, c4_w, c4_b, t4, 4, 256, 386, 48, 64, 512, 512, 0, 3, 3, 2, 1, 24, 32, 1, 0);
    launch_conv(stream, t4, c41_w, c41_b, cat4, 4, 512, 512, 24, 32, 512, 770, 0, 3, 3, 1, 1, 24, 32, 1, 0);   // o4
    launch_conv(stream, cat4, c5_w, c5_b, t5, 4, 512, 770, 24, 32, 512, 512, 0, 3, 3, 2, 1, 12, 16, 1, 0);
    launch_conv(stream, t5, c51_w, c51_b, cat5, 4, 512, 512, 12, 16, 512, 1026, 0, 3, 3, 1, 1, 12, 16, 1, 0);  // o5
    launch_conv(stream, cat5, c6_w, c6_b, t6, 4, 512, 1026, 12, 16, 1024, 1024, 0, 3, 3, 2, 1, 6, 8, 1, 0);
    launch_conv(stream, t6, c61_w, c61_b, o6, 4, 1024, 1024, 6, 8, 1024, 1024, 0, 3, 3, 1, 1, 6, 8, 1, 0);

    // --- expanding path ---
    launch_conv(stream, o6, pf6_w, pf6_b, flow6, 4, 1024, 1024, 6, 8, 2, 2, 0, 3, 3, 1, 1, 6, 8, 0, 0);
    launch_conv(stream, o6, d5_w, d5_b, cat5, 4, 1024, 1024, 6, 8, 512, 1026, 512, 4, 4, 1, 2, 12, 16, 1, 1);
    launch_conv(stream, flow6, u6_w, u6_b, cat5, 4, 2, 2, 6, 8, 2, 1026, 1024, 4, 4, 1, 2, 12, 16, 0, 1);
    launch_conv(stream, cat5, pf5_w, pf5_b, flow5, 4, 1026, 1026, 12, 16, 2, 2, 0, 3, 3, 1, 1, 12, 16, 0, 0);

    launch_conv(stream, cat5, d4_w, d4_b, cat4, 4, 1026, 1026, 12, 16, 256, 770, 512, 4, 4, 1, 2, 24, 32, 1, 1);
    launch_conv(stream, flow5, u5_w, u5_b, cat4, 4, 2, 2, 12, 16, 2, 770, 768, 4, 4, 1, 2, 24, 32, 0, 1);
    launch_conv(stream, cat4, pf4_w, pf4_b, flow4, 4, 770, 770, 24, 32, 2, 2, 0, 3, 3, 1, 1, 24, 32, 0, 0);

    launch_conv(stream, cat4, d3_w, d3_b, cat3, 4, 770, 770, 24, 32, 128, 386, 256, 4, 4, 1, 2, 48, 64, 1, 1);
    launch_conv(stream, flow4, u4_w, u4_b, cat3, 4, 2, 2, 24, 32, 2, 386, 384, 4, 4, 1, 2, 48, 64, 0, 1);
    launch_conv(stream, cat3, pf3_w, pf3_b, flow3, 4, 386, 386, 48, 64, 2, 2, 0, 3, 3, 1, 1, 48, 64, 0, 0);

    launch_conv(stream, cat3, d2_w, d2_b, cat2, 4, 386, 386, 48, 64, 64, 194, 128, 4, 4, 1, 2, 96, 128, 1, 1);
    launch_conv(stream, flow3, u3_w, u3_b, cat2, 4, 2, 2, 48, 64, 2, 194, 192, 4, 4, 1, 2, 96, 128, 0, 1);
    launch_conv(stream, cat2, pf2_w, pf2_b, flow2, 4, 194, 194, 96, 128, 2, 2, 0, 3, 3, 1, 1, 96, 128, 0, 0);

    // --- final x4 bilinear upsample, scaled by 20 ---
    {
        const int total = 4 * 2 * 384 * 512;
        upsample_kernel<<<(total + 255) / 256, 256, 0, stream>>>(
            flow2, (float*)d_out, 4, 2, 96, 128, 384, 512, 20.0f);
    }
}
